// StandardLayer_9474697855295
// MI455X (gfx1250) — matrix operand order repaired and checked by reading, not compiled, not run on hardware
//
#include <hip/hip_runtime.h>
#include <hip/hip_bf16.h>
#include <math.h>
#include <stdint.h>

// ---------------------------------------------------------------------------
// Transformer encoder layer (pre-LN, MHA + GELU MLP), B=4 S=2048 D=1024 H=16
// HD=64 F=4096, fp32 in/out, bf16 WMMA internals (V_WMMA_F32_16X16X32_BF16).
// GEMM A-tiles staged with GLOBAL_LOAD_ASYNC_TO_LDS_B128 (double-buffered);
// weights pre-transposed so B fragments stream straight from global/L2.
// ---------------------------------------------------------------------------

typedef __bf16 bf16_t;
typedef __bf16 v16bf __attribute__((ext_vector_type(16)));
typedef float  v8f   __attribute__((ext_vector_type(8)));

constexpr int CB = 4, CS = 2048, CD = 1024, CH = 16, CHD = 64, CF = 4096;

__device__ __forceinline__ v8f wmma_bf16(v16bf a, v16bf b, v8f c) {
  // D = A(16x32 bf16) * B(32x16 bf16) + C(16x16 f32)
  return __builtin_amdgcn_wmma_f32_16x16x32_bf16(
      /*neg_a=*/false, a, /*neg_b=*/false, b,
      /*c_mod=*/(short)0, c, /*reuse_a=*/false, /*reuse_b=*/false);
}

// A fragment: 16x32 bf16, element (m,k) at base[m*ld + k].
// Lanes 0-15: m=lane, K=0..15; lanes 16-31: same m, K=16..31.
__device__ __forceinline__ v16bf frag_a(const bf16_t* base, int ld) {
  const int t  = threadIdx.x & 31;
  const int m  = t & 15, hk = t >> 4;
  const bf16_t* p = base + m * ld;
  v16bf a;
#pragma unroll
  for (int j = 0; j < 4; ++j) {
    int k = 16 * hk + 2 * j;
    a[2 * j]     = p[k];
    a[2 * j + 1] = p[k + 1];
  }
#pragma unroll
  for (int j = 0; j < 4; ++j) {
    int k = 16 * hk + 8 + 2 * j;
    a[8 + 2 * j]     = p[k];
    a[8 + 2 * j + 1] = p[k + 1];
  }
  return a;
}

// B fragment: 32x16 bf16, element (k,n) at base[n*ld + k] (K contiguous,
// 32B per lane -> global_load_b128 pairs). Lanes 0-15: K=0..15, 16-31: 16..31.
__device__ __forceinline__ v16bf frag_b(const bf16_t* base, int ld) {
  const int t  = threadIdx.x & 31;
  const int n  = t & 15, hk = t >> 4;
  const bf16_t* p = base + n * ld + 16 * hk;
  v16bf b;
#pragma unroll
  for (int j = 0; j < 8; ++j) {
    b[2 * j]     = p[2 * j];
    b[2 * j + 1] = p[2 * j + 1];
  }
  return b;
}

// ---------------------------------------------------------------------------
// fp32 [K][N] -> bf16 transposed [N][K] (tiled LDS transpose)
// grid: (N/32, K/32), 256 threads (32 x 8)
// ---------------------------------------------------------------------------
__global__ __launch_bounds__(256) void cvt_t_bf16_kernel(
    const float* __restrict__ in, bf16_t* __restrict__ out, int Kd, int Nd) {
  __shared__ float t[32][33];
  const int n0 = blockIdx.x * 32, k0 = blockIdx.y * 32;
  const int tx = threadIdx.x & 31, ty = threadIdx.x >> 5;
#pragma unroll
  for (int i = ty; i < 32; i += 8)
    t[i][tx] = in[(size_t)(k0 + i) * Nd + n0 + tx];
  __syncthreads();
#pragma unroll
  for (int i = ty; i < 32; i += 8)
    out[(size_t)(n0 + i) * Kd + k0 + tx] = (bf16_t)t[tx][i];
}

// ---------------------------------------------------------------------------
// LayerNorm: one row (D=1024) per block, fp32 in -> bf16 out
// ---------------------------------------------------------------------------
__global__ __launch_bounds__(256) void ln_kernel(const float* __restrict__ x,
                                                 const float* __restrict__ g,
                                                 const float* __restrict__ be,
                                                 bf16_t* __restrict__ out) {
  __shared__ float s1[256], s2[256];
  const int tid = threadIdx.x;
  const float* row = x + (size_t)blockIdx.x * CD;
  float sum = 0.f, sq = 0.f;
#pragma unroll
  for (int i = tid; i < CD; i += 256) {
    float v = row[i];
    sum += v;
    sq  += v * v;
  }
  s1[tid] = sum;
  s2[tid] = sq;
  __syncthreads();
  for (int off = 128; off > 0; off >>= 1) {
    if (tid < off) {
      s1[tid] += s1[tid + off];
      s2[tid] += s2[tid + off];
    }
    __syncthreads();
  }
  const float mean = s1[0] * (1.0f / CD);
  const float var  = s2[0] * (1.0f / CD) - mean * mean;
  const float rstd = rsqrtf(var + 1e-5f);
  bf16_t* orow = out + (size_t)blockIdx.x * CD;
#pragma unroll
  for (int i = tid; i < CD; i += 256)
    orow[i] = (bf16_t)((row[i] - mean) * rstd * g[i] + be[i]);
}

// ---------------------------------------------------------------------------
// Tiled BF16 WMMA GEMM: C[M,N] = A[M,K] * Wt[N,K]^T + bias
//   mode 0: Q  -> bf16 [B,H,S,HD]
//   mode 1: K  -> bf16 [B,H,S,HD]
//   mode 2: V  -> bf16 [B,H,HD,S]   (transposed for attn B-fragments)
//   mode 3: f32_out = f32_res + val            (O-proj residual)
//   mode 4: bf_out = gelu(val)                 (MLP up)
//   mode 5: f32_out = f32_res + val            (MLP down -> d_out)
// Block tile 128(M) x 64(N), K-step 32, 8 waves (4x2) of 32x32 each.
// A tile: async global->LDS b128, double-buffered. B: direct from global Wt.
// ---------------------------------------------------------------------------
#define LDA 72  // 144B row pitch: 16B aligned, 36-dword bank stride

__global__ __launch_bounds__(256) void gemm_bf16_kernel(
    const bf16_t* __restrict__ A, const bf16_t* __restrict__ Wt,
    const float* __restrict__ bias, int M, int N, int K, int mode,
    bf16_t* __restrict__ bf_out, float* __restrict__ f32_out,
    const float* __restrict__ f32_res) {
  (void)M; (void)N;
  __shared__ __align__(16) bf16_t As[2][128][LDA];  // [m][k] row-major

  const int tid  = threadIdx.x;
  const int w    = tid >> 5, lane = tid & 31;
  const int nl   = lane & 15, hi = lane >> 4;
  const int wm   = w & 3, wn = w >> 2;
  const int bm0  = blockIdx.y * 128, bn0 = blockIdx.x * 64;

  // 512 16B-chunks of the 128x32 A tile; 2 async copies per thread
  const int c0r = (tid * 2) >> 3;          // chunk row for i=0 (pairs share row)
  const int c0c = ((tid * 2) & 7) * 8;     // col offset in elements (i=0)
  const int c1c = c0c + 8;                 // i=1 chunk is next 8 elements

  auto stage_a = [&](int buf, int k0) {
    const bf16_t* g0 = A + (size_t)(bm0 + c0r) * K + k0 + c0c;
    const bf16_t* g1 = A + (size_t)(bm0 + c0r) * K + k0 + c1c;
    uint32_t l0 = (uint32_t)(size_t)&As[buf][c0r][c0c];
    uint32_t l1 = (uint32_t)(size_t)&As[buf][c0r][c1c];
    asm volatile("global_load_async_to_lds_b128 %0, %1, off"
                 :: "v"(l0), "v"((uint64_t)(size_t)g0) : "memory");
    asm volatile("global_load_async_to_lds_b128 %0, %1, off"
                 :: "v"(l1), "v"((uint64_t)(size_t)g1) : "memory");
  };

  v8f acc[2][2] = {};
  stage_a(0, 0);

  const bf16_t* wt0 = Wt + (size_t)(bn0 + wn * 32) * K;       // sn = 0
  const bf16_t* wt1 = Wt + (size_t)(bn0 + wn * 32 + 16) * K;  // sn = 1

  int buf = 0;
  for (int k0 = 0; k0 < K; k0 += 32) {
    asm volatile("s_wait_asynccnt 0x0" ::: "memory");
    __syncthreads();
    if (k0 + 32 < K) stage_a(buf ^ 1, k0 + 32);  // overlap next-slab copy

    v16bf a0 = frag_a(&As[buf][wm * 32][0], LDA);
    v16bf a1 = frag_a(&As[buf][wm * 32 + 16][0], LDA);
    v16bf b0 = frag_b(wt0 + k0, K);
    v16bf b1 = frag_b(wt1 + k0, K);
    acc[0][0] = wmma_bf16(a0, b0, acc[0][0]);
    acc[0][1] = wmma_bf16(a0, b1, acc[0][1]);
    acc[1][0] = wmma_bf16(a1, b0, acc[1][0]);
    acc[1][1] = wmma_bf16(a1, b1, acc[1][1]);
    buf ^= 1;
  }

  // Epilogue. C layout: lane L, vgpr r -> (M = r + 8*(L>>4), N = L&15)
#pragma unroll
  for (int sm = 0; sm < 2; ++sm)
#pragma unroll
    for (int sn = 0; sn < 2; ++sn)
#pragma unroll
      for (int r = 0; r < 8; ++r) {
        int row = bm0 + wm * 32 + sm * 16 + r + 8 * hi;
        int col = bn0 + wn * 32 + sn * 16 + nl;
        float val = acc[sm][sn][r] + bias[col];
        if (mode <= 2) {
          int b_ = row >> 11;        // row / S
          int s_ = row & (CS - 1);
          int h_ = col >> 6;         // col / HD
          int hd = col & (CHD - 1);
          if (mode == 2)  // V transposed [B,H,HD,S]
            bf_out[(((size_t)b_ * CH + h_) * CHD + hd) * CS + s_] = (bf16_t)val;
          else            // Q / K [B,H,S,HD]
            bf_out[(((size_t)b_ * CH + h_) * CS + s_) * CHD + hd] = (bf16_t)val;
        } else if (mode == 4) {
          float g_ = 0.5f * val * (1.0f + erff(val * 0.70710678118654752f));
          bf_out[(size_t)row * N + col] = (bf16_t)g_;
        } else {  // mode 3 / 5: residual add, f32 out
          size_t idx = (size_t)row * N + col;
          f32_out[idx] = f32_res[idx] + val;
        }
      }
}

// ---------------------------------------------------------------------------
// Flash attention: grid (S/64, H, B), 128 threads = 4 waves; each wave owns a
// 16-row Q tile and streams keys in chunks of 32 with online softmax.
// q,k: [B,H,S,HD] bf16;  v: [B,H,HD,S] bf16;  out: [B,S,D] bf16
// ---------------------------------------------------------------------------
__global__ __launch_bounds__(128) void attn_kernel(
    const bf16_t* __restrict__ qb, const bf16_t* __restrict__ kb,
    const bf16_t* __restrict__ vt, bf16_t* __restrict__ out) {
  const int tid  = threadIdx.x;
  const int w    = tid >> 5, lane = tid & 31;
  const int nl   = lane & 15, hi = lane >> 4;
  const int h    = blockIdx.y, b = blockIdx.z;
  const int q0   = blockIdx.x * 64 + w * 16;
  const size_t bh = (size_t)b * CH + h;
  const bf16_t* qp = qb + bh * (size_t)CS * CHD;
  const bf16_t* kp = kb + bh * (size_t)CS * CHD;
  const bf16_t* vp = vt + bh * (size_t)CHD * CS;

  __shared__ float  pf[4][16][32];   // f32 score tile per wave
  __shared__ bf16_t pb[4][16][32];   // bf16 P tile (A-fragment source)
  __shared__ float  rm[4][16], rl[4][16], rc[4][16];

  if (lane < 16) {
    rm[w][lane] = -1e30f;
    rl[w][lane] = 0.f;
  }

  const v16bf aq0 = frag_a(qp + (size_t)q0 * CHD, CHD);       // d = 0..31
  const v16bf aq1 = frag_a(qp + (size_t)q0 * CHD + 32, CHD);  // d = 32..63

  v8f o[4] = {};
  const float scale = 0.125f;  // 1/sqrt(HD)

  for (int kc = 0; kc < CS; kc += 32) {
    // scores(16x32) = Q(16x64) * K^T ; B element (k=d, n) = K[kc+n][d]
    v8f c0 = {}, c1 = {};
    {
      v16bf bk = frag_b(kp + (size_t)kc * CHD, CHD);
      c0 = wmma_bf16(aq0, bk, c0);
      bk = frag_b(kp + (size_t)kc * CHD + 32, CHD);
      c0 = wmma_bf16(aq1, bk, c0);
      bk = frag_b(kp + (size_t)(kc + 16) * CHD, CHD);
      c1 = wmma_bf16(aq0, bk, c1);
      bk = frag_b(kp + (size_t)(kc + 16) * CHD + 32, CHD);
      c1 = wmma_bf16(aq1, bk, c1);
    }
    __syncthreads();
#pragma unroll
    for (int r = 0; r < 8; ++r) {
      pf[w][r + 8 * hi][nl]      = c0[r] * scale;
      pf[w][r + 8 * hi][16 + nl] = c1[r] * scale;
    }
    __syncthreads();
    if (lane < 16) {  // one lane per Q row: online softmax over this chunk
      const int m = lane;
      float mo = rm[w][m];
      float tmax = mo;
#pragma unroll
      for (int j = 0; j < 32; ++j) tmax = fmaxf(tmax, pf[w][m][j]);
      float corr = expf(mo - tmax);
      float sum = 0.f;
#pragma unroll
      for (int j = 0; j < 32; ++j) {
        float e = expf(pf[w][m][j] - tmax);
        sum += e;
        pb[w][m][j] = (bf16_t)e;
      }
      rm[w][m] = tmax;
      rl[w][m] = rl[w][m] * corr + sum;
      rc[w][m] = corr;
    }
    __syncthreads();
    // rescale running O by correction factor of each row
#pragma unroll
    for (int r = 0; r < 8; ++r) {
      float corr = rc[w][r + 8 * hi];
#pragma unroll
      for (int nt = 0; nt < 4; ++nt) o[nt][r] *= corr;
    }
    // O(16x64) += P(16x32) * V(32x64); B element (k, n=d) = V^T[d][kc+k]
    v16bf ap = frag_a(&pb[w][0][0], 32);
#pragma unroll
    for (int nt = 0; nt < 4; ++nt) {
      v16bf bv = frag_b(vp + (size_t)(nt * 16) * CS + kc, CS);
      o[nt] = wmma_bf16(ap, bv, o[nt]);
    }
  }

  // normalize by row sum and write bf16 [B,S,D]
#pragma unroll
  for (int r = 0; r < 8; ++r) {
    const int m   = r + 8 * hi;
    const float il = 1.0f / rl[w][m];
    const size_t row = (size_t)b * CS + (q0 + m);
#pragma unroll
    for (int nt = 0; nt < 4; ++nt)
      out[row * CD + h * CHD + nt * 16 + nl] = (bf16_t)(o[nt][r] * il);
  }
}

// ---------------------------------------------------------------------------
// Host-side orchestration
// ---------------------------------------------------------------------------
extern "C" void kernel_launch(void* const* d_in, const int* in_sizes, int n_in,
                              void* d_out, int out_size, void* d_ws,
                              size_t ws_size, hipStream_t stream) {
  (void)in_sizes; (void)n_in; (void)out_size; (void)ws_size;
  const float* x   = (const float*)d_in[0];
  const float* Wq  = (const float*)d_in[1];
  const float* bq  = (const float*)d_in[2];
  const float* Wk  = (const float*)d_in[3];
  const float* bk  = (const float*)d_in[4];
  const float* Wv  = (const float*)d_in[5];
  const float* bv  = (const float*)d_in[6];
  const float* Wo  = (const float*)d_in[7];
  const float* bo  = (const float*)d_in[8];
  const float* W1  = (const float*)d_in[9];
  const float* b1  = (const float*)d_in[10];
  const float* W2  = (const float*)d_in[11];
  const float* b2  = (const float*)d_in[12];
  const float* g1  = (const float*)d_in[13];
  const float* be1 = (const float*)d_in[14];
  const float* g2  = (const float*)d_in[15];
  const float* be2 = (const float*)d_in[16];

  char* ws = (char*)d_ws;
  size_t off = 0;
  auto alloc = [&](size_t bytes) -> void* {
    void* p = ws + off;
    off += (bytes + 255) & ~(size_t)255;
    return p;
  };
  constexpr size_t BSr = (size_t)CB * CS;  // 8192 rows

  bf16_t* wq_t   = (bf16_t*)alloc((size_t)CD * CD * 2);  // [N][K] transposed
  bf16_t* wk_t   = (bf16_t*)alloc((size_t)CD * CD * 2);
  bf16_t* wv_t   = (bf16_t*)alloc((size_t)CD * CD * 2);
  bf16_t* wo_t   = (bf16_t*)alloc((size_t)CD * CD * 2);
  bf16_t* w1_t   = (bf16_t*)alloc((size_t)CD * CF * 2);  // [F][D]
  bf16_t* w2_t   = (bf16_t*)alloc((size_t)CF * CD * 2);  // [D][F]
  bf16_t* xn_bf  = (bf16_t*)alloc(BSr * CD * 2);
  bf16_t* q_bf   = (bf16_t*)alloc(BSr * CD * 2);
  bf16_t* k_bf   = (bf16_t*)alloc(BSr * CD * 2);
  bf16_t* vt_bf  = (bf16_t*)alloc(BSr * CD * 2);
  bf16_t* ao_bf  = (bf16_t*)alloc(BSr * CD * 2);
  float*  x1     = (float*)alloc(BSr * CD * 4);
  bf16_t* xn2_bf = (bf16_t*)alloc(BSr * CD * 2);
  bf16_t* h_bf   = (bf16_t*)alloc(BSr * CF * 2);

  // 1) weights fp32 [K][N] -> bf16 transposed [N][K]
  dim3 t_dd(CD / 32, CD / 32);
  cvt_t_bf16_kernel<<<t_dd, 256, 0, stream>>>(Wq, wq_t, CD, CD);
  cvt_t_bf16_kernel<<<t_dd, 256, 0, stream>>>(Wk, wk_t, CD, CD);
  cvt_t_bf16_kernel<<<t_dd, 256, 0, stream>>>(Wv, wv_t, CD, CD);
  cvt_t_bf16_kernel<<<t_dd, 256, 0, stream>>>(Wo, wo_t, CD, CD);
  cvt_t_bf16_kernel<<<dim3(CF / 32, CD / 32), 256, 0, stream>>>(W1, w1_t, CD, CF);
  cvt_t_bf16_kernel<<<dim3(CD / 32, CF / 32), 256, 0, stream>>>(W2, w2_t, CF, CD);

  // 2) LN1
  ln_kernel<<<(int)BSr, 256, 0, stream>>>(x, g1, be1, xn_bf);

  // 3) QKV projections
  dim3 g_dd(CD / 64, (int)BSr / 128);
  gemm_bf16_kernel<<<g_dd, 256, 0, stream>>>(xn_bf, wq_t, bq, (int)BSr, CD, CD,
                                             0, q_bf, nullptr, nullptr);
  gemm_bf16_kernel<<<g_dd, 256, 0, stream>>>(xn_bf, wk_t, bk, (int)BSr, CD, CD,
                                             1, k_bf, nullptr, nullptr);
  gemm_bf16_kernel<<<g_dd, 256, 0, stream>>>(xn_bf, wv_t, bv, (int)BSr, CD, CD,
                                             2, vt_bf, nullptr, nullptr);

  // 4) attention
  attn_kernel<<<dim3(CS / 64, CH, CB), 128, 0, stream>>>(q_bf, k_bf, vt_bf,
                                                         ao_bf);

  // 5) O projection + residual
  gemm_bf16_kernel<<<g_dd, 256, 0, stream>>>(ao_bf, wo_t, bo, (int)BSr, CD, CD,
                                             3, nullptr, x1, x);

  // 6) LN2
  ln_kernel<<<(int)BSr, 256, 0, stream>>>(x1, g2, be2, xn2_bf);

  // 7) MLP up + GELU
  dim3 g_df(CF / 64, (int)BSr / 128);
  gemm_bf16_kernel<<<g_df, 256, 0, stream>>>(xn2_bf, w1_t, b1, (int)BSr, CF,
                                             CD, 4, h_bf, nullptr, nullptr);

  // 8) MLP down + residual -> d_out (fp32)
  gemm_bf16_kernel<<<g_dd, 256, 0, stream>>>(h_bf, w2_t, b2, (int)BSr, CD, CF,
                                             5, nullptr, (float*)d_out, x1);
}
